// FoveatedAttention_90993177133359
// MI455X (gfx1250) — matrix operand order repaired and checked by reading, not compiled, not run on hardware
//
#include <hip/hip_runtime.h>
#include <math.h>

// ---------------------------------------------------------------------------
// FoveatedAttention on MI455X (gfx1250, wave32).
//   * all big GEMMs (convs as implicit GEMM, bott, cp1..3, ctx) use
//     v_wmma_f32_16x16x32_bf16, f32 accumulate (bf16 operands are fine:
//     every WMMA output feeds a LayerNorm).
//   * each wave computes a 64x16 output strip: 4 WMMAs share one B fragment,
//     so the 269 MB cp1_w stream is fetched exactly once grid-wide.
//   * cp1/bott stream B tiles via the Tensor Data Mover into LDS
//     (tensor_load_to_lds + s_wait_tensorcnt, double buffered).
// ---------------------------------------------------------------------------

typedef __attribute__((ext_vector_type(16))) __bf16 v16bf;
typedef __attribute__((ext_vector_type(8)))  float  v8f;
typedef __attribute__((ext_vector_type(4)))  unsigned int u32x4;
typedef __attribute__((ext_vector_type(8)))  int    i32x8;
typedef __attribute__((ext_vector_type(4)))  int    i32x4;

__device__ __forceinline__ v8f wmma_bf16(v16bf a, v16bf b, v8f c) {
  return __builtin_amdgcn_wmma_f32_16x16x32_bf16(false, a, false, b, (short)0, c,
                                                 false, false);
}

// 16-bit A fragment 16x32 as built here (not the ISA table order): lane holds row M=lane&15; for
// lane-half h the fragment's VGPR v holds K pair:
//   K = h*16 + 2v,(+1)  (contiguous order, matching the B fragments)
// K pairs are consecutive -> load as float2 from a row pointer.
__device__ __forceinline__ v16bf frag_a_row(const float2* p2, int h, float scale) {
  v16bf af;
#pragma unroll
  for (int v = 0; v < 8; ++v) {
    int k0 = h * 16 + 2 * v;
    float2 d = p2[k0 >> 1];
    af[2 * v]     = (__bf16)(d.x * scale);
    af[2 * v + 1] = (__bf16)(d.y * scale);
  }
  return af;
}

// B-matrix 32x16: lane holds col N=lane&15; lanes 0-15 K=0..15, 16-31 K=16..31.
// p points at B[kb][n0+col]; ldn = row stride in elements.
__device__ __forceinline__ v16bf frag_b_glob(const float* p, int ldn, int h) {
  v16bf bf;
#pragma unroll
  for (int e = 0; e < 16; ++e)
    bf[e] = (__bf16)p[(size_t)(h * 16 + e) * ldn];
  return bf;
}

__device__ __forceinline__ v16bf frag_b_lds(const float* lds, int h, int col) {
  v16bf bf;
#pragma unroll
  for (int e = 0; e < 16; ++e)
    bf[e] = (__bf16)lds[(h * 16 + e) * 16 + col];
  return bf;
}

// ---------------------------------------------------------------------------
// Implicit-GEMM conv, 'SAME':  Y = sin(conv(X,W)+bias) [+ res]
// X:[B,Lin,Cin] W:[taps,Cin,Cout] Y:[B,Lout,Cout]. Cin mult of 32, Lout of 64.
// One wave: 64 output positions x 16 cout (4 WMMAs reuse one weight fragment).
// K loop split taps x (Cin/32) so a 32-K block sits inside one tap: position
// validity is one compare per tap, OOB handled by a 0/1 fragment scale.
// ---------------------------------------------------------------------------
__global__ void conv_sin_res_wmma(const float* __restrict__ X,
                                  const float* __restrict__ Wt,
                                  const float* __restrict__ bias,
                                  const float* __restrict__ res,
                                  float* __restrict__ Y,
                                  int Bn, int Lin, int Cin,
                                  int Lout, int Cout, int taps, int stride, int pad) {
  int wid = threadIdx.x >> 5, lane = threadIdx.x & 31;
  int ntiles = Cout >> 4;
  int mquads = (Bn * Lout) >> 6;
  int tile = blockIdx.x * (blockDim.x >> 5) + wid;
  if (tile >= mquads * ntiles) return;            // wave-uniform exit
  int nt = tile % ntiles, mq = tile / ntiles;
  int qPerB = Lout >> 6;
  int b   = mq / qPerB;
  int lo0 = (mq % qPerB) << 6;
  int co0 = nt << 4;
  int h = lane >> 4, m = lane & 15, col = m;
  const float* Xb = X + (size_t)b * Lin * Cin;
  v8f acc[4] = {{}, {}, {}, {}};
  for (int t = 0; t < taps; ++t) {
    const float* rowp[4];
    float fm[4];
#pragma unroll
    for (int s = 0; s < 4; ++s) {
      int p  = lo0 + (s * 16 + m) * stride + t - pad;
      int pc = p < 0 ? 0 : (p >= Lin ? Lin - 1 : p);
      fm[s]   = (p == pc) ? 1.f : 0.f;
      rowp[s] = Xb + (size_t)pc * Cin;
    }
    const float* wcol = Wt + (size_t)t * Cin * Cout + co0 + col;
    for (int cb = 0; cb < Cin; cb += 32) {
      v16bf bf = frag_b_glob(wcol + (size_t)cb * Cout, Cout, h);
#pragma unroll
      for (int s = 0; s < 4; ++s) {
        v16bf af = frag_a_row((const float2*)(rowp[s] + cb), h, fm[s]);
        acc[s] = wmma_bf16(af, bf, acc[s]);
      }
    }
  }
  // C/D layout: col = lane&15, rows (lane>>4)*8 .. +7 across 8 acc VGPRs.
  float bv = bias[co0 + col];
  int rb = h * 8;
#pragma unroll
  for (int s = 0; s < 4; ++s)
#pragma unroll
    for (int r = 0; r < 8; ++r) {
      int lo = lo0 + s * 16 + rb + r;
      size_t idx = ((size_t)b * Lout + lo) * Cout + co0 + col;
      float v = __builtin_sinf(acc[s][r] + bv);
      if (res) v += res[idx];
      Y[idx] = v;
    }
}

// ---------------------------------------------------------------------------
// Batched GEMM, M fixed = 64:  Y[z] = sin(A[z] @ B[z] + bias[z])
// One wave = whole 64-row strip of one 16-col tile (B fragment reused 4x).
// ---------------------------------------------------------------------------
__global__ void gemm_sin_wmma(const float* __restrict__ A, const float* __restrict__ Bw,
                              const float* __restrict__ bias, float* __restrict__ Y,
                              int N, int K,
                              long long sA, long long sB, long long sBias, long long sY) {
  const float* Ab  = A    + (size_t)blockIdx.z * sA;
  const float* Bb  = Bw   + (size_t)blockIdx.z * sB;
  const float* bsb = bias + (size_t)blockIdx.z * sBias;
  float*       Yb  = Y    + (size_t)blockIdx.z * sY;
  int wid = threadIdx.x >> 5, lane = threadIdx.x & 31;
  int nt = blockIdx.x * (blockDim.x >> 5) + wid;
  if (nt >= (N >> 4)) return;
  int n0 = nt << 4;
  int h = lane >> 4, m = lane & 15, col = m;
  const float* arow[4];
#pragma unroll
  for (int s = 0; s < 4; ++s) arow[s] = Ab + (size_t)(s * 16 + m) * K;
  const float* bp = Bb + n0 + col;
  v8f acc[4] = {{}, {}, {}, {}};
  for (int kb = 0; kb < K; kb += 32) {
    if (kb + 64 < K)  // stream-prefetch next weight block
      __builtin_prefetch(Bb + (size_t)(kb + 64) * N + n0, 0, 1);
    v16bf bf = frag_b_glob(bp + (size_t)kb * N, N, h);
#pragma unroll
    for (int s = 0; s < 4; ++s) {
      v16bf af = frag_a_row((const float2*)(arow[s] + kb), h, 1.f);
      acc[s] = wmma_bf16(af, bf, acc[s]);
    }
  }
  float bv = bsb[n0 + col];
  int rb = h * 8;
#pragma unroll
  for (int s = 0; s < 4; ++s)
#pragma unroll
    for (int r = 0; r < 8; ++r)
      Yb[(size_t)(s * 16 + rb + r) * N + n0 + col] = __builtin_sinf(acc[s][r] + bv);
}

// ---------------------------------------------------------------------------
// TDM variant: B tiles (32xK rows, 16 cols) streamed into LDS by the Tensor
// Data Mover, double buffered, synchronized with s_wait_tensorcnt.
// ---------------------------------------------------------------------------
#if defined(__AMDGCN__) && __has_builtin(__builtin_amdgcn_tensor_load_to_lds)
#define FOV_HAVE_TDM 1
__device__ __forceinline__ void tdm_load_b_tile(unsigned ldsOff, const float* gsrc,
                                                int rowStrideElems) {
  unsigned long long ga = (unsigned long long)(size_t)gsrc;
  // D# group0: count=1, lds_addr, global_addr(56:0), type=2 ("image")
  u32x4 g0 = { 1u, ldsOff, (unsigned)ga,
               (unsigned)((ga >> 32) & 0x01FFFFFFu) | 0x80000000u };
  // D# group1: data_size=4B; tensor dims huge (no clipping);
  // tile 16(x) x 32(y); dim0 stride = rowStrideElems.
  i32x8 g1 = { 0x20000,                 // data_size=2 (4B) at bits 17:16
               0,                        // dim0 low16 (of 0x40000000) = 0
               0x4000,                   // dim0 hi16 | dim1 low16
               0x00104000,               // dim1 hi16 | tile_dim0=16
               32,                       // tile_dim1=32, tile_dim2=0
               rowStrideElems,           // dim0_stride low32
               0, 0 };
  i32x4 z4 = { 0, 0, 0, 0 };
#if __clang_major__ >= 23
  i32x8 z8 = { 0, 0, 0, 0, 0, 0, 0, 0 };
  __builtin_amdgcn_tensor_load_to_lds(g0, g1, z4, z4, z8, 0);
#else
  __builtin_amdgcn_tensor_load_to_lds(g0, g1, z4, z4, 0);
#endif
}
#else
#define FOV_HAVE_TDM 0
#endif

__global__ void gemm_sin_wmma_tdm(const float* __restrict__ A, const float* __restrict__ Bw,
                                  const float* __restrict__ bias, float* __restrict__ Y,
                                  int N, int K,
                                  long long sA, long long sB, long long sBias, long long sY) {
#if FOV_HAVE_TDM
  __shared__ float ldsB[8][2][32 * 16];     // per-wave double buffer (32 KB total)
  const float* Ab  = A    + (size_t)blockIdx.z * sA;
  const float* Bb  = Bw   + (size_t)blockIdx.z * sB;
  const float* bsb = bias + (size_t)blockIdx.z * sBias;
  float*       Yb  = Y    + (size_t)blockIdx.z * sY;
  int wid = threadIdx.x >> 5, lane = threadIdx.x & 31;
  int nt = blockIdx.x * (blockDim.x >> 5) + wid;
  if (nt >= (N >> 4)) return;
  int n0 = nt << 4;
  int h = lane >> 4, m = lane & 15, col = m;
  const float* arow[4];
#pragma unroll
  for (int s = 0; s < 4; ++s) arow[s] = Ab + (size_t)(s * 16 + m) * K;
  unsigned ldsOff[2];
  ldsOff[0] = (unsigned)(size_t)(void*)&ldsB[wid][0][0];
  ldsOff[1] = (unsigned)(size_t)(void*)&ldsB[wid][1][0];
  v8f acc[4] = {{}, {}, {}, {}};
  tdm_load_b_tile(ldsOff[0], Bb + n0, N);
  int buf = 0, kb = 0;
  for (; kb + 32 < K; kb += 32) {
    tdm_load_b_tile(ldsOff[buf ^ 1], Bb + (size_t)(kb + 32) * N + n0, N);
    __builtin_amdgcn_s_wait_tensorcnt((short)1);    // current tile landed
    v16bf bf = frag_b_lds(&ldsB[wid][buf][0], h, col);
#pragma unroll
    for (int s = 0; s < 4; ++s) {
      v16bf af = frag_a_row((const float2*)(arow[s] + kb), h, 1.f);
      acc[s] = wmma_bf16(af, bf, acc[s]);
    }
    buf ^= 1;
  }
  __builtin_amdgcn_s_wait_tensorcnt((short)0);      // last tile
  {
    v16bf bf = frag_b_lds(&ldsB[wid][buf][0], h, col);
#pragma unroll
    for (int s = 0; s < 4; ++s) {
      v16bf af = frag_a_row((const float2*)(arow[s] + kb), h, 1.f);
      acc[s] = wmma_bf16(af, bf, acc[s]);
    }
  }
  float bv = bsb[n0 + col];
  int rb = h * 8;
#pragma unroll
  for (int s = 0; s < 4; ++s)
#pragma unroll
    for (int r = 0; r < 8; ++r)
      Yb[(size_t)(s * 16 + rb + r) * N + n0 + col] = __builtin_sinf(acc[s][r] + bv);
#else
  // Fallback: identical math with direct global B fragments.
  const float* Ab  = A    + (size_t)blockIdx.z * sA;
  const float* Bb  = Bw   + (size_t)blockIdx.z * sB;
  const float* bsb = bias + (size_t)blockIdx.z * sBias;
  float*       Yb  = Y    + (size_t)blockIdx.z * sY;
  int wid = threadIdx.x >> 5, lane = threadIdx.x & 31;
  int nt = blockIdx.x * (blockDim.x >> 5) + wid;
  if (nt >= (N >> 4)) return;
  int n0 = nt << 4;
  int h = lane >> 4, m = lane & 15, col = m;
  const float* arow[4];
#pragma unroll
  for (int s = 0; s < 4; ++s) arow[s] = Ab + (size_t)(s * 16 + m) * K;
  v8f acc[4] = {{}, {}, {}, {}};
  for (int kb = 0; kb < K; kb += 32) {
    v16bf bf = frag_b_glob(Bb + (size_t)kb * N + n0 + col, N, h);
#pragma unroll
    for (int s = 0; s < 4; ++s) {
      v16bf af = frag_a_row((const float2*)(arow[s] + kb), h, 1.f);
      acc[s] = wmma_bf16(af, bf, acc[s]);
    }
  }
  float bv = bsb[n0 + col];
  int rb = h * 8;
#pragma unroll
  for (int s = 0; s < 4; ++s)
#pragma unroll
    for (int r = 0; r < 8; ++r)
      Yb[(size_t)(s * 16 + rb + r) * N + n0 + col] = __builtin_sinf(acc[s][r] + bv);
#endif
}

// ---------------------------------------------------------------------------
// In-place LayerNorm over last dim (Cl in {64,128,256}). 1 wave per row.
// ---------------------------------------------------------------------------
__global__ void layernorm_rows(float* __restrict__ Y, const float* __restrict__ g,
                               const float* __restrict__ bta, int rows, int Cl,
                               int nIter, int rowsPerG) {
  int wid = threadIdx.x >> 5, lane = threadIdx.x & 31;
  int row = blockIdx.x * (blockDim.x >> 5) + wid;
  if (row >= rows) return;
  const float* gp = g   + (size_t)(row / rowsPerG) * Cl;
  const float* bp = bta + (size_t)(row / rowsPerG) * Cl;
  float* yp = Y + (size_t)row * Cl;
  float rv[8];
  float s = 0.f, s2 = 0.f;
  for (int i = 0; i < nIter; ++i) {
    float v = yp[lane + 32 * i];
    rv[i] = v; s += v; s2 += v * v;
  }
  for (int off = 16; off > 0; off >>= 1) {
    s  += __shfl_xor(s,  off, 32);
    s2 += __shfl_xor(s2, off, 32);
  }
  float inv  = 1.f / (float)Cl;
  float mean = s * inv;
  float var  = s2 * inv - mean * mean;
  float rstd = rsqrtf(var + 1e-3f);
  for (int i = 0; i < nIter; ++i) {
    int c = lane + 32 * i;
    yp[c] = (rv[i] - mean) * rstd * gp[c] + bp[c];
  }
}

// eff_bias[n,f] = cp1_b[n,f] + sum_{j=32..63} cp1_w[n, G+j, f]   (posenc(0))
__global__ void eff_bias_kernel(const float* __restrict__ cp1w,
                                const float* __restrict__ cp1b,
                                float* __restrict__ eb) {
  int idx = blockIdx.x * blockDim.x + threadIdx.x;
  if (idx >= 8 * 256) return;
  int n = idx >> 8, f = idx & 255;
  const size_t GC = 32768 + 64;
  float s = cp1b[idx];
  for (int j = 32; j < 64; ++j)
    s += cp1w[((size_t)n * GC + 32768 + j) * 256 + f];
  eb[idx] = s;
}

// centers[n,b] = sigmoid( dot(hln[n,b,:], cp_end_w[n,:,0]) + cp_end_b[n] )
__global__ void centers_kernel(const float* __restrict__ hln, const float* __restrict__ ew,
                               const float* __restrict__ ebia, float* __restrict__ centers,
                               float* __restrict__ outCenters) {
  int wid = threadIdx.x >> 5, lane = threadIdx.x & 31;
  int nb = blockIdx.x * (blockDim.x >> 5) + wid;
  if (nb >= 8 * 64) return;
  int n = nb >> 6;
  const float* hp = hln + (size_t)nb * 256;
  const float* wp = ew  + (size_t)n  * 256;
  float s = 0.f;
  for (int i = 0; i < 8; ++i) s += hp[lane + 32 * i] * wp[lane + 32 * i];
  for (int off = 16; off > 0; off >>= 1) s += __shfl_xor(s, off, 32);
  float d = s + ebia[n];
  float c = 1.f / (1.f + __builtin_expf(-d));
  if (lane == 0) { centers[nb] = c; outCenters[nb] = c; }
}

// mp[n,b,0:64]=window mean of x around center; mp[n,b,64:128]=posenc(center)
__global__ void mp_kernel(const float* __restrict__ X, const float* __restrict__ centers,
                          float* __restrict__ mp) {
  int wid = threadIdx.x >> 5, lane = threadIdx.x & 31;
  int nb = blockIdx.x * (blockDim.x >> 5) + wid;
  if (nb >= 8 * 64) return;
  int b = nb & 63;
  float c = centers[nb];
  int cp = (int)(c * 2048.f);
  int lo = cp - 32; if (lo < 0) lo = 0;
  int hi = cp + 32; if (hi > 2048) hi = 2048;
  int cnt = hi - lo; if (cnt < 1) cnt = 1;
  float invc = 1.f / (float)cnt;
  float* mpp = mp + (size_t)nb * 128;
#pragma unroll
  for (int half = 0; half < 2; ++half) {
    int ch = lane + 32 * half;
    float s = 0.f;
    for (int r = lo; r < hi; ++r)
      s += X[((size_t)b * 2048 + r) * 64 + ch];
    mpp[ch] = s * invc;
  }
  float dv = __builtin_expf((float)(2 * lane) * (-9.210340371976184f / 64.f));
  mpp[64 + lane]      = __builtin_sinf(c * dv);
  mpp[64 + 32 + lane] = __builtin_cosf(c * dv);
}

// final[b,f] = cc[b,f] + sum_n feat[n,b,f] / sqrt(8)
__global__ void final_kernel(const float* __restrict__ cc, const float* __restrict__ feat,
                             float* __restrict__ out) {
  int idx = blockIdx.x * blockDim.x + threadIdx.x;
  if (idx >= 64 * 256) return;
  float s = 0.f;
  for (int n = 0; n < 8; ++n) s += feat[(size_t)n * 64 * 256 + idx];
  out[idx] = cc[idx] + s * 0.35355339059327373f;
}

// ---------------------------------------------------------------------------
extern "C" void kernel_launch(void* const* d_in, const int* in_sizes, int n_in,
                              void* d_out, int out_size, void* d_ws, size_t ws_size,
                              hipStream_t stream) {
  const float* x         = (const float*)d_in[0];
  const float* at_full_w = (const float*)d_in[1];
  const float* at_full_b = (const float*)d_in[2];
  const float* to4_w     = (const float*)d_in[3];
  const float* to4_b     = (const float*)d_in[4];
  const float* at4_w     = (const float*)d_in[5];
  const float* at4_b     = (const float*)d_in[6];
  const float* to16_w    = (const float*)d_in[7];
  const float* to16_b    = (const float*)d_in[8];
  const float* at16_w    = (const float*)d_in[9];
  const float* at16_b    = (const float*)d_in[10];
  const float* ln_full_g = (const float*)d_in[11];
  const float* ln_full_b = (const float*)d_in[12];
  const float* ln_4_g    = (const float*)d_in[13];
  const float* ln_4_b    = (const float*)d_in[14];
  const float* ln_16_g   = (const float*)d_in[15];
  const float* ln_16_b   = (const float*)d_in[16];
  const float* bott_w    = (const float*)d_in[17];
  const float* bott_b    = (const float*)d_in[18];
  const float* emb_w     = (const float*)d_in[19];
  const float* emb_b     = (const float*)d_in[20];
  const float* ln_cc_g   = (const float*)d_in[21];
  const float* ln_cc_b   = (const float*)d_in[22];
  const float* cp1_w     = (const float*)d_in[23];
  const float* cp1_b     = (const float*)d_in[24];
  const float* cp2_w     = (const float*)d_in[25];
  const float* cp2_b     = (const float*)d_in[26];
  const float* cp3_w     = (const float*)d_in[27];
  const float* cp3_b     = (const float*)d_in[28];
  const float* cp_ln_g   = (const float*)d_in[29];
  const float* cp_ln_b   = (const float*)d_in[30];
  const float* cp_end_w  = (const float*)d_in[31];
  const float* cp_end_b  = (const float*)d_in[32];
  const float* ctx_w     = (const float*)d_in[33];
  const float* ctx_b     = (const float*)d_in[34];
  const float* ctx_ln_g  = (const float*)d_in[35];
  const float* ctx_ln_b  = (const float*)d_in[36];

  float* ws   = (float*)d_ws;
  float* full = ws;  ws += (size_t)64 * 2048 * 64;
  float* t4   = ws;  ws += (size_t)64 * 512 * 128;
  float* o4   = ws;  ws += (size_t)64 * 512 * 128;
  float* t16  = ws;  ws += (size_t)64 * 128 * 256;
  float* o16  = ws;  ws += (size_t)64 * 128 * 256;   // == gc [64, 32768]
  float* cc1  = ws;  ws += 64 * 64;
  float* cc   = ws;  ws += 64 * 256;
  float* h1   = ws;  ws += 8 * 64 * 256;
  float* h2   = ws;  ws += 8 * 64 * 256;
  float* eb   = ws;  ws += 8 * 256;
  float* cent = ws;  ws += 8 * 64;
  float* mp   = ws;  ws += 8 * 64 * 128;
  float* feat = ws;  ws += 8 * 64 * 256;

  float* out_final = (float*)d_out;
  float* out_cent  = out_final + 64 * 256;

  const int TPB = 256, WPB = 8;

  // full = LN( sin(conv5(x)) + x )
  int nt = (64 * 2048 / 64) * (64 / 16);
  conv_sin_res_wmma<<<(nt + WPB - 1) / WPB, TPB, 0, stream>>>(
      x, at_full_w, at_full_b, x, full, 64, 2048, 64, 2048, 64, 5, 1, 2);
  layernorm_rows<<<(64 * 2048 + WPB - 1) / WPB, TPB, 0, stream>>>(
      full, ln_full_g, ln_full_b, 64 * 2048, 64, 2, 64 * 2048);

  // t4 = sin(conv8_s4(full))
  nt = (64 * 512 / 64) * (128 / 16);
  conv_sin_res_wmma<<<(nt + WPB - 1) / WPB, TPB, 0, stream>>>(
      full, to4_w, to4_b, nullptr, t4, 64, 2048, 64, 512, 128, 8, 4, 2);

  // o4 = LN( sin(conv4(t4)) + t4 )
  conv_sin_res_wmma<<<(nt + WPB - 1) / WPB, TPB, 0, stream>>>(
      t4, at4_w, at4_b, t4, o4, 64, 512, 128, 512, 128, 4, 1, 1);
  layernorm_rows<<<(64 * 512 + WPB - 1) / WPB, TPB, 0, stream>>>(
      o4, ln_4_g, ln_4_b, 64 * 512, 128, 4, 64 * 512);

  // t16 = sin(conv8_s4(o4))
  nt = (64 * 128 / 64) * (256 / 16);
  conv_sin_res_wmma<<<(nt + WPB - 1) / WPB, TPB, 0, stream>>>(
      o4, to16_w, to16_b, nullptr, t16, 64, 512, 128, 128, 256, 8, 4, 2);

  // o16 = LN( sin(conv4(t16)) + t16 )  -> gc
  conv_sin_res_wmma<<<(nt + WPB - 1) / WPB, TPB, 0, stream>>>(
      t16, at16_w, at16_b, t16, o16, 64, 128, 256, 128, 256, 4, 1, 1);
  layernorm_rows<<<(64 * 128 + WPB - 1) / WPB, TPB, 0, stream>>>(
      o16, ln_16_g, ln_16_b, 64 * 128, 256, 8, 64 * 128);

  // cc1 = sin(gc @ bott_w + b)  — TDM-streamed B
  gemm_sin_wmma_tdm<<<dim3(1, 1, 1), TPB, 0, stream>>>(
      o16, bott_w, bott_b, cc1, 64, 32768, 0, 0, 0, 0);
  // cc = LN( sin(cc1 @ emb_w + b) )
  gemm_sin_wmma<<<dim3(2, 1, 1), TPB, 0, stream>>>(
      cc1, emb_w, emb_b, cc, 256, 64, 0, 0, 0, 0);
  layernorm_rows<<<8, TPB, 0, stream>>>(cc, ln_cc_g, ln_cc_b, 64, 256, 8, 64);

  // fold posenc(0) into cp1 bias; h1 = sin(gc @ cp1_w[:, :G] + eff_bias) — TDM
  eff_bias_kernel<<<8, TPB, 0, stream>>>(cp1_w, cp1_b, eb);
  gemm_sin_wmma_tdm<<<dim3(2, 1, 8), TPB, 0, stream>>>(
      o16, cp1_w, eb, h1, 256, 32768,
      0, (long long)(32768 + 64) * 256, 256, 64 * 256);
  // h2 = sin(h1 @ cp2_w + b);  h1 = sin(h2 @ cp3_w + b)
  gemm_sin_wmma<<<dim3(2, 1, 8), TPB, 0, stream>>>(
      h1, cp2_w, cp2_b, h2, 256, 256, 64 * 256, 256 * 256, 256, 64 * 256);
  gemm_sin_wmma<<<dim3(2, 1, 8), TPB, 0, stream>>>(
      h2, cp3_w, cp3_b, h1, 256, 256, 64 * 256, 256 * 256, 256, 64 * 256);
  layernorm_rows<<<(512 + WPB - 1) / WPB, TPB, 0, stream>>>(
      h1, cp_ln_g, cp_ln_b, 512, 256, 8, 64);

  // centers (also written to output tail)
  centers_kernel<<<64, TPB, 0, stream>>>(h1, cp_end_w, cp_end_b, cent, out_cent);

  // mp = [window-mean(x), posenc(center)]
  mp_kernel<<<64, TPB, 0, stream>>>(x, cent, mp);

  // feat = LN( sin(mp @ ctx_w + ctx_b) )
  gemm_sin_wmma<<<dim3(2, 1, 8), TPB, 0, stream>>>(
      mp, ctx_w, ctx_b, feat, 256, 128, 64 * 128, 128 * 256, 256, 64 * 256);
  layernorm_rows<<<(512 + WPB - 1) / WPB, TPB, 0, stream>>>(
      feat, ctx_ln_g, ctx_ln_b, 512, 256, 8, 64);

  // final = cc + sum_n feat / sqrt(8)
  final_kernel<<<64, TPB, 0, stream>>>(cc, feat, out_final);
}
